// MLPBlock_16028817949441
// MI455X (gfx1250) — matrix operand order repaired and checked by reading, not compiled, not run on hardware
//
#include <hip/hip_runtime.h>
#include <hip/hip_bf16.h>

// Problem sizes (fixed by the reference)
#define B_  4
#define S_  2048
#define H_  1024
#define I_  1024
#define E_  8
#define K_  2
#define T_  (B_ * S_)           // 8192 tokens
#define LIMIT_ 7.0f
#define ALPHA_ 1.702f
#define EPS_   1e-5f

#define AB_STRIDE 1032          // 1024 + 8 bf16 pad (keeps 16B alignment, staggers banks)
#define MTILE 64                // tokens per expert block (4x B-operand reuse)

typedef __bf16 v16bf __attribute__((ext_vector_type(16)));
typedef float  v8f   __attribute__((ext_vector_type(8)));

union FragAB { uint4 q[2]; v16bf v; };

// ---- scalar bf16 helpers (RNE via hip_bf16.h) ----
static __device__ __forceinline__ unsigned short f2bf_bits(float f) {
    union { __hip_bfloat16 h; unsigned short u; } cv;
    cv.h = __float2bfloat16(f);
    return cv.u;
}
static __device__ __forceinline__ float bfbits2f(unsigned short u) {
    union { __hip_bfloat16 h; unsigned short u; } cv;
    cv.u = u;
    return __bfloat162float(cv.h);
}

// -------------------------------------------------------------------------
// Kernel 0: zero the accumulator / counters / lists region of the workspace
// -------------------------------------------------------------------------
__global__ void zero_kernel(unsigned int* __restrict__ p, size_t n) {
    size_t i = (size_t)blockIdx.x * blockDim.x + threadIdx.x;
    size_t stride = (size_t)gridDim.x * blockDim.x;
    for (; i < n; i += stride) p[i] = 0u;
}

// -------------------------------------------------------------------------
// Kernel 1: RMSNorm (bf16 round-trip semantics) + router top-2 + softmax.
// One block (256 thr = 8 waves) per token. Wave w computes expert w's logit.
// -------------------------------------------------------------------------
__global__ __launch_bounds__(256) void rmsnorm_router_kernel(
    const float* __restrict__ x, const float* __restrict__ normw,
    const unsigned short* __restrict__ gatew, const unsigned short* __restrict__ gateb,
    unsigned short* __restrict__ nf,
    int* __restrict__ counts, int* __restrict__ lists, float* __restrict__ gates)
{
    const int token = blockIdx.x;
    const int tid   = threadIdx.x;
    __shared__ float red[256];
    __shared__ unsigned short nfs[H_];
    __shared__ float logit_s[E_];

    const float* xr = x + (size_t)token * H_;
    float xv[4];
    float ss = 0.f;
#pragma unroll
    for (int i = 0; i < 4; ++i) {
        int h = tid + 256 * i;
        float v = bfbits2f(f2bf_bits(xr[h]));   // xc = bf16(x), xf = f32(xc)
        xv[i] = v;
        ss += v * v;
    }
    red[tid] = ss;
    __syncthreads();
    for (int off = 128; off > 0; off >>= 1) {
        if (tid < off) red[tid] += red[tid + off];
        __syncthreads();
    }
    const float scale = rsqrtf(red[0] / (float)H_ + EPS_);
#pragma unroll
    for (int i = 0; i < 4; ++i) {
        int h = tid + 256 * i;
        unsigned short nb = f2bf_bits(xv[i] * scale * normw[h]);
        nfs[h] = nb;
        nf[(size_t)token * H_ + h] = nb;
    }
    __syncthreads();

    // router: wave w -> expert w (fp32 accum over bf16 operands)
    const int wave = tid >> 5, lane = tid & 31;
    const unsigned short* gw = gatew + wave * H_;
    float d = 0.f;
    for (int i = 0; i < 32; ++i) {
        int h = lane + 32 * i;
        d += bfbits2f(nfs[h]) * bfbits2f(gw[h]);
    }
    red[tid] = d;
    __syncthreads();
    for (int off = 16; off > 0; off >>= 1) {
        if (lane < off) red[tid] += red[tid + off];
        __syncthreads();
    }
    if (lane == 0)
        logit_s[wave] = bfbits2f(f2bf_bits(red[tid] + bfbits2f(gateb[wave]))); // bf16 round-trip
    __syncthreads();

    if (tid == 0) {
        int i0 = 0; float s0 = logit_s[0];
        for (int e = 1; e < E_; ++e)
            if (logit_s[e] > s0) { s0 = logit_s[e]; i0 = e; }
        int i1 = -1; float s1 = -1e30f;
        for (int e = 0; e < E_; ++e) {
            if (e == i0) continue;
            if (logit_s[e] > s1) { s1 = logit_s[e]; i1 = e; }
        }
        float e1 = __expf(s1 - s0);
        float denom = 1.f + e1;
        float w0 = 1.f / denom, w1 = e1 / denom;
        int p0 = atomicAdd(&counts[i0], 1);
        lists[i0 * T_ + p0] = token; gates[i0 * T_ + p0] = w0;
        int p1 = atomicAdd(&counts[i1], 1);
        lists[i1 * T_ + p1] = token; gates[i1 * T_ + p1] = w1;
    }
}

// -------------------------------------------------------------------------
// Kernel 2: per-(expert, 64-token tile) fused MLP1 + SwiGLU + MLP2 with
// v_wmma_f32_16x16x32_bf16.  grid = E * (T/64), 512 threads = 16 waves.
// Each B fragment (weight slice) feeds FOUR A sub-tiles (rows 0-15, 16-31,
// 32-47, 48-63): 8 WMMAs per pair of B-row loads -> 8 B/lane/WMMA of weight
// traffic, ~1.6 GB aggregate out of L2 for the whole dispatch.
// Dynamic LDS: abuf (64x1032 bf16) + actbuf (same) = 264 KB (< 320 KB/WG).
// -------------------------------------------------------------------------
__global__ __launch_bounds__(512) void moe_expert_kernel(
    const unsigned short* __restrict__ nf,
    const int* __restrict__ counts,
    const int* __restrict__ lists,
    const float* __restrict__ gates,
    const unsigned short* __restrict__ w1,
    const unsigned short* __restrict__ b1,
    const unsigned short* __restrict__ w2,
    const unsigned short* __restrict__ b2,
    float* __restrict__ acc)
{
    const int e    = blockIdx.x >> 7;       // / 128 tiles
    const int tile = blockIdx.x & 127;
    const int cnt  = counts[e];
    if (tile * MTILE >= cnt) return;        // uniform across block

    extern __shared__ __attribute__((aligned(16))) unsigned short smem[];
    unsigned short* abuf   = smem;                      // 64 x 1032 bf16 = 132 KB
    unsigned short* actbuf = smem + MTILE * AB_STRIDE;  // 64 x 1032 bf16 = 132 KB
    __shared__ int   toks[MTILE];
    __shared__ float gws[MTILE];

    const int tid  = threadIdx.x;
    const int lane = tid & 31;
    const int wave = tid >> 5;              // 0..15
    const int halfsel = lane >> 4;          // 0: K low halves, 1: K high halves
    const int ln      = lane & 15;          // row (A) / column (B,C,D)

    if (tid < MTILE) {
        int slot = tile * MTILE + tid;
        if (slot < cnt) {
            toks[tid] = lists[e * T_ + slot];
            gws[tid]  = gates[e * T_ + slot];
        } else { toks[tid] = 0; gws[tid] = 0.f; }   // padded rows: weight 0
    }
    __syncthreads();

    // Stage normed activations: 64 gathered token rows x 1024 bf16 into LDS.
    {
        int r = tid >> 3, c = tid & 7;      // 8 threads per row, 16 uint4 each
        const uint4* src = reinterpret_cast<const uint4*>(nf + (size_t)toks[r] * H_);
#pragma unroll
        for (int s = 0; s < 16; ++s) {
            int q = s * 8 + c;              // 128 uint4 per row
            *reinterpret_cast<uint4*>(&abuf[r * AB_STRIDE + q * 8]) = src[q];
        }
    }
    __syncthreads();

    // A fragment (16x32 bf16) from LDS buffer `buf`, this lane's row = rowbase (callers pass ln + 16*s):
    // lanes 0-15 hold halves {k0..k0+7, k0+16..23}, lanes 16-31 hold
    // {k0+8..15, k0+24..31} (ISA 7.12.2).
    auto loadA = [&](const unsigned short* buf, int rowbase, int k0) -> v16bf {
        FragAB f;
        const unsigned short* p = &buf[rowbase * AB_STRIDE + k0 + (halfsel << 3)];
        f.q[0] = *reinterpret_cast<const uint4*>(p);
        f.q[1] = *reinterpret_cast<const uint4*>(p + 16);
        return f.v;
    };
    // B fragment (32x16 bf16): lane = column; lanes 0-15 hold K={k0..k0+7, k0+16..23},
    // lanes 16-31 hold K={k0+8..15, k0+24..31} — same K order as A.
    auto loadB = [&](const unsigned short* row, int k0) -> v16bf {
        FragAB f;
        const unsigned short* p = row + k0 + (halfsel << 3);
        f.q[0] = *reinterpret_cast<const uint4*>(p);
        f.q[1] = *reinterpret_cast<const uint4*>(p + 16);
        return f.v;
    };

    // ---------------- MLP1 + clipped SwiGLU ----------------
    const size_t w1off = (size_t)e * (2 * I_) * H_;
#pragma unroll 1
    for (int t = 0; t < 4; ++t) {           // 16 waves x 4 = 64 I-tiles
        const int col = (wave * 4 + t) * 16 + ln;          // I-column index
        const unsigned short* rowg = w1 + w1off + (size_t)(2 * col) * H_;     // even: glu
        const unsigned short* rowl = w1 + w1off + (size_t)(2 * col + 1) * H_; // odd: linear
        v8f cg[4], cl[4];
#pragma unroll
        for (int s = 0; s < 4; ++s) {
            cg[s] = (v8f){0.f,0.f,0.f,0.f,0.f,0.f,0.f,0.f};
            cl[s] = (v8f){0.f,0.f,0.f,0.f,0.f,0.f,0.f,0.f};
        }
        for (int kb = 0; kb < 32; ++kb) {
            const int k0 = kb * 32;
            v16bf bg = loadB(rowg, k0);
            v16bf bl = loadB(rowl, k0);
#pragma unroll
            for (int s = 0; s < 4; ++s) {
                v16bf a = loadA(abuf, ln + 16 * s, k0);
                cg[s] = __builtin_amdgcn_wmma_f32_16x16x32_bf16(false, a, false, bg, (short)0, cg[s], false, false);
                cl[s] = __builtin_amdgcn_wmma_f32_16x16x32_bf16(false, a, false, bl, (short)0, cl[s], false, false);
            }
        }
        const float bgb = bfbits2f(b1[e * 2 * I_ + 2 * col]);
        const float blb = bfbits2f(b1[e * 2 * I_ + 2 * col + 1]);
#pragma unroll
        for (int s = 0; s < 4; ++s) {
#pragma unroll
            for (int j = 0; j < 8; ++j) {
                const int row = j + (halfsel << 3) + 16 * s;   // C/D layout row
                float g  = fminf(cg[s][j] + bgb, LIMIT_);
                float l  = fminf(fmaxf(cl[s][j] + blb, -LIMIT_), LIMIT_);
                float sg = 1.0f / (1.0f + __expf(-ALPHA_ * g));
                actbuf[row * AB_STRIDE + col] = f2bf_bits(g * sg * (l + 1.0f));
            }
        }
    }
    __syncthreads();                    // all act tiles written before MLP2 reads

    // ---------------- MLP2 + gated scatter-add ----------------
    const size_t w2off = (size_t)e * H_ * I_;
#pragma unroll 1
    for (int t = 0; t < 4; ++t) {           // 16 waves x 4 = 64 H-tiles
        const int hcol = (wave * 4 + t) * 16 + ln;         // H-column index
        const unsigned short* row2 = w2 + w2off + (size_t)hcol * I_;
        v8f c[4];
#pragma unroll
        for (int s = 0; s < 4; ++s)
            c[s] = (v8f){0.f,0.f,0.f,0.f,0.f,0.f,0.f,0.f};
        for (int kb = 0; kb < 32; ++kb) {
            const int k0 = kb * 32;
            v16bf b = loadB(row2, k0);
#pragma unroll
            for (int s = 0; s < 4; ++s) {
                v16bf a = loadA(actbuf, ln + 16 * s, k0);
                c[s] = __builtin_amdgcn_wmma_f32_16x16x32_bf16(false, a, false, b, (short)0, c[s], false, false);
            }
        }
        const float b2f = bfbits2f(b2[e * H_ + hcol]);
#pragma unroll
        for (int s = 0; s < 4; ++s) {
#pragma unroll
            for (int j = 0; j < 8; ++j) {
                const int row = j + (halfsel << 3) + 16 * s;
                atomicAdd(&acc[(size_t)toks[row] * H_ + hcol],
                          gws[row] * (c[s][j] + b2f));
            }
        }
    }
}

// -------------------------------------------------------------------------
// Kernel 3: out = f32( bf16( bf16(x) + bf16(acc) ) )   (output dtype = f32)
// -------------------------------------------------------------------------
__global__ __launch_bounds__(256) void finalize_kernel(
    const float* __restrict__ x, const float* __restrict__ acc, float* __restrict__ out)
{
    size_t i = (size_t)blockIdx.x * blockDim.x + threadIdx.x;
    float xs = bfbits2f(f2bf_bits(x[i]));
    float cs = bfbits2f(f2bf_bits(acc[i]));
    out[i] = bfbits2f(f2bf_bits(xs + cs));
}

// -------------------------------------------------------------------------
extern "C" void kernel_launch(void* const* d_in, const int* in_sizes, int n_in,
                              void* d_out, int out_size, void* d_ws, size_t ws_size,
                              hipStream_t stream)
{
    (void)in_sizes; (void)n_in; (void)out_size; (void)ws_size;
    const float*          x     = (const float*)d_in[0];
    const float*          normw = (const float*)d_in[1];
    const unsigned short* gatew = (const unsigned short*)d_in[2];   // bf16 bits
    const unsigned short* gateb = (const unsigned short*)d_in[3];
    const unsigned short* w1    = (const unsigned short*)d_in[4];
    const unsigned short* b1    = (const unsigned short*)d_in[5];
    const unsigned short* w2    = (const unsigned short*)d_in[6];
    const unsigned short* b2    = (const unsigned short*)d_in[7];
    float* out = (float*)d_out;

    // Workspace layout (~51 MB):
    //   acc    : T*H  f32   (33.5 MB)
    //   counts : E    i32   (padded to 64 B)
    //   lists  : E*T  i32   (256 KB)
    //   gates  : E*T  f32   (256 KB)
    //   nf     : T*H  bf16  (16.8 MB)
    char* ws = (char*)d_ws;
    float* acc = (float*)ws;
    size_t off = (size_t)T_ * H_ * sizeof(float);
    int*   counts = (int*)(ws + off);   off += 64;
    int*   lists  = (int*)(ws + off);   off += (size_t)E_ * T_ * sizeof(int);
    float* gates  = (float*)(ws + off); off += (size_t)E_ * T_ * sizeof(float);
    unsigned short* nf = (unsigned short*)(ws + off);

    // acc + counts + lists + gates are contiguous dwords from ws start
    const size_t zero_dw = (size_t)T_ * H_ + 16 + (size_t)E_ * T_ * 2;
    zero_kernel<<<2048, 256, 0, stream>>>((unsigned int*)ws, zero_dw);

    rmsnorm_router_kernel<<<T_, 256, 0, stream>>>(x, normw, gatew, gateb,
                                                  nf, counts, lists, gates);

    const size_t lds_bytes = (size_t)2 * MTILE * AB_STRIDE * sizeof(unsigned short); // 264 KB
    moe_expert_kernel<<<E_ * (T_ / MTILE), 512, lds_bytes, stream>>>(nf, counts, lists, gates,
                                                                     w1, b1, w2, b2, acc);

    finalize_kernel<<<(T_ * H_) / 256, 256, 0, stream>>>(x, acc, out);
}
